// MHGNN_56126632624863
// MI455X (gfx1250) — matrix operand order repaired and checked by reading, not compiled, not run on hardware
//
#include <hip/hip_runtime.h>
#include <hip/hip_bf16.h>

typedef _Float16 half_t;
typedef __attribute__((ext_vector_type(16))) _Float16 v16h;
typedef __attribute__((ext_vector_type(8)))  float    v8f;

struct __align__(16) U8 { unsigned int u[8]; };   // 32 bytes = one v16h fragment

static __device__ __forceinline__ unsigned int pack2(float x, float y) {
    _Float16 lo = (_Float16)x, hi = (_Float16)y;
    unsigned int a = (unsigned int)__builtin_bit_cast(unsigned short, lo);
    unsigned int b = (unsigned int)__builtin_bit_cast(unsigned short, hi);
    return a | (b << 16);
}

// CDNA5 packed-f16 atomic add (2 channels per atomic op, device scope -> L2 atomic units)
static __device__ __forceinline__ void atomic_pk_add_f16(half_t* addr, unsigned int data) {
    asm volatile("global_atomic_pk_add_f16 %0, %1, off scope:SCOPE_DEV"
                 :: "v"(addr), "v"(data) : "memory");
}

#define NNODES 4096
#define NEDGES 16384
#define NROWS  262144          // B*N
#define FSTRIDE 416            // padded MID (400 -> 416 = 13*32)

// ---------------- init: x0 -> feat[:,0:16] f16, zero pad cols, zero accum ----
__global__ void gnn_init_kernel(const float* __restrict__ x0,
                                half_t* __restrict__ feat,
                                float* __restrict__ accum) {
    int r = blockIdx.x * 256 + threadIdx.x;
    if (r < NROWS) {
        const float* src = x0 + (size_t)r * 16;
        half_t* dst = feat + (size_t)r * FSTRIDE;
        #pragma unroll
        for (int i = 0; i < 16; i += 2)
            *(unsigned int*)(dst + i) = pack2(src[i], src[i + 1]);
        U8 z = {};
        *(U8*)(dst + 400) = z;  // 16 zero halves (cols 400..415)
    }
    if (blockIdx.x == 0 && threadIdx.x < 64) accum[threadIdx.x] = 0.f;
}

// ---------------- fold BN running stats + c1 bias into scale/shift -----------
__global__ void gnn_bnprep_kernel(const float* c1b, const float* g, const float* b,
                                  const float* mean, const float* var,
                                  const float* c2wsrc,
                                  float* s, float* t, float* w) {
    int i = threadIdx.x;  // 256
    float rs = rsqrtf(var[i] + 1e-5f);
    float sc = g[i] * rs;
    s[i] = sc;
    t[i] = (c1b[i] - mean[i]) * sc + b[i];
    w[i] = c2wsrc[i];
}

// ---- pack W[Kreal][Ncols] (f32) into WMMA B-fragment layout, K zero-padded --
// layout: pk[((nt*KK + kk)*32 + lane)*8 + v] = half2(W[k][n], W[k+1][n])
__global__ void gnn_packb_kernel(const float* __restrict__ W,
                                 unsigned int* __restrict__ pk,
                                 int KK, int Kreal, int Ncols) {
    int idx = blockIdx.x * 256 + threadIdx.x;
    int total = (Ncols / 16) * KK * 256;
    if (idx >= total) return;
    int v = idx & 7;
    int lane = (idx >> 3) & 31;
    int kk = (idx >> 8) % KK;
    int nt = (idx >> 8) / KK;
    int n = nt * 16 + (lane & 15);
    int k = kk * 32 + 2 * (v & 3) + 16 * (v >> 2) + 8 * (lane >> 4);
    float w0 = (k     < Kreal) ? W[(size_t)k * Ncols + n]       : 0.f;
    float w1 = (k + 1 < Kreal) ? W[(size_t)(k + 1) * Ncols + n] : 0.f;
    pk[idx] = pack2(w0, w1);
}

// ---------------- zero scratch ----------------------------------------------
__global__ void gnn_zero_kernel(float4* __restrict__ p, long n4) {
    long i = (long)blockIdx.x * 256 + threadIdx.x;
    if (i < n4) { float4 z; z.x = z.y = z.z = z.w = 0.f; p[i] = z; }
}

// ------- layer-1 scatter: f32 source, F=16, f16 agg stride 32, pk atomics ----
__global__ void gnn_scatter16_kernel(const float* __restrict__ x,
                                     const int* __restrict__ links,
                                     half_t* __restrict__ agg) {
    long idx = (long)blockIdx.x * 256 + threadIdx.x;   // B*E*2 threads
    int h8 = (int)(idx & 1) * 8;
    int e  = (int)((idx >> 1) & (NEDGES - 1));
    int b  = (int)(idx >> 15);                          // /(E*2)
    if (b >= 64) return;
    int src = links[(size_t)b * 2 * NEDGES + e];
    int dst = links[(size_t)b * 2 * NEDGES + NEDGES + e];
    const float* xr = x + ((size_t)b * NNODES + src) * 16 + h8;
    float4 f0 = *(const float4*)xr;
    float4 f1 = *(const float4*)(xr + 4);
    half_t* ar = agg + ((size_t)b * NNODES + dst) * 32 + h8;
    atomic_pk_add_f16(ar + 0, pack2(f0.x, f0.y));
    atomic_pk_add_f16(ar + 2, pack2(f0.z, f0.w));
    atomic_pk_add_f16(ar + 4, pack2(f1.x, f1.y));
    atomic_pk_add_f16(ar + 6, pack2(f1.z, f1.w));
}

// ------- layer-2/3 scatter: f16 feat slice, F=128, f16 agg, pk atomics -------
__global__ void gnn_scatter128_kernel(const half_t* __restrict__ feat, int featOff,
                                      const int* __restrict__ links,
                                      half_t* __restrict__ agg) {
    long idx = (long)blockIdx.x * 256 + threadIdx.x;   // B*E*16 threads
    int c8 = (int)(idx & 15) * 8;
    int e  = (int)((idx >> 4) & (NEDGES - 1));
    int b  = (int)(idx >> 18);                          // /(E*16)
    if (b >= 64) return;
    int src = links[(size_t)b * 2 * NEDGES + e];
    int dst = links[(size_t)b * 2 * NEDGES + NEDGES + e];
    const half_t* xr = feat + ((size_t)b * NNODES + src) * FSTRIDE + featOff + c8;
    uint4 u = *(const uint4*)xr;                        // 8 packed halves, no convert
    half_t* ar = agg + ((size_t)b * NNODES + dst) * 128 + c8;
    atomic_pk_add_f16(ar + 0, u.x);
    atomic_pk_add_f16(ar + 2, u.y);
    atomic_pk_add_f16(ar + 4, u.z);
    atomic_pk_add_f16(ar + 6, u.w);
}

// ------- fused WMMA GEMM (agg f16 [rows][KK*32] @ packed W) + bias + LN + ReLU
// 8 waves/block, wave = one 16-row M-tile, 8 N-tiles of 16 -> 128 cols.
// Packed W staged in LDS once per block -> ds_load_b128 B fragments.
template <int KK>
__global__ __launch_bounds__(256)
void gnn_gemm_ln_kernel(const half_t* __restrict__ agg,
                        const uint4* __restrict__ pkW,   // KK*512 uint4
                        const float* __restrict__ bias,
                        const float* __restrict__ lng, const float* __restrict__ lnb,
                        half_t* __restrict__ feat, int featOff) {
    __shared__ float dscr[8 * 16 * 132];                // padded rows: no bank conflicts
    __shared__ uint4 wlds[512 * KK];                    // packed W fragments
    const int lane   = threadIdx.x & 31;
    const int wave   = threadIdx.x >> 5;
    const int khalf  = lane >> 4;
    const int rowBase = blockIdx.x * 128 + wave * 16;
    const int kpad   = KK * 32;

    for (int i = threadIdx.x; i < 512 * KK; i += 256) wlds[i] = pkW[i];

    // Preload all A fragments: two b128 loads each (pairs are contiguous in f16 agg)
    const uint4* ap = (const uint4*)(agg + (size_t)(rowBase + (lane & 15)) * kpad);
    v16h afr[KK];
    #pragma unroll
    for (int kk = 0; kk < KK; ++kk) {
        U8 t;
        *(uint4*)&t.u[0] = ap[kk * 4 + khalf];          // K pairs 0..7  (or 8..15)
        *(uint4*)&t.u[4] = ap[kk * 4 + 2 + khalf];      // K pairs 16..23 (or 24..31)
        afr[kk] = __builtin_bit_cast(v16h, t);
    }
    __syncthreads();

    float* wd = dscr + wave * (16 * 132);
    for (int nt = 0; nt < 8; ++nt) {
        v8f c = {};
        #pragma unroll
        for (int kk = 0; kk < KK; ++kk) {
            const uint4* bp = wlds + ((nt * KK + kk) * 32 + lane) * 2;
            U8 bu;
            *(uint4*)&bu.u[0] = bp[0];
            *(uint4*)&bu.u[4] = bp[1];
            v16h bfr = __builtin_bit_cast(v16h, bu);
            c = __builtin_amdgcn_wmma_f32_16x16x32_f16(
                    false, afr[kk], false, bfr, (short)0, c, false, false);
        }
        int n = nt * 16 + (lane & 15);
        float bb = bias[n];
        #pragma unroll
        for (int i = 0; i < 8; ++i)
            wd[(i + 8 * khalf) * 132 + n] = c[i] + bb;
    }
    __syncthreads();

    if (lane < 16) {                                    // one lane per row: LN + ReLU
        const float* row = wd + lane * 132;
        float s = 0.f, s2 = 0.f;
        for (int ci = 0; ci < 128; ++ci) { float v = row[ci]; s += v; s2 += v * v; }
        float mu = s * (1.f / 128.f);
        float var = s2 * (1.f / 128.f) - mu * mu;
        float rs = rsqrtf(var + 1e-5f);
        half_t* out = feat + (size_t)(rowBase + lane) * FSTRIDE + featOff;
        for (int ci = 0; ci < 128; ci += 2) {
            float v0 = (row[ci]     - mu) * rs * lng[ci]     + lnb[ci];
            float v1 = (row[ci + 1] - mu) * rs * lng[ci + 1] + lnb[ci + 1];
            v0 = v0 > 0.f ? v0 : 0.f;
            v1 = v1 > 0.f ? v1 : 0.f;
            *(unsigned int*)(out + ci) = pack2(v0, v1);
        }
    }
}

// ------- critic: WMMA feat[rows][416] @ c1(416x256 packed) + BN + ReLU + c2 dot
__global__ __launch_bounds__(256)
void gnn_critic_kernel(const half_t* __restrict__ feat,
                       const unsigned int* __restrict__ pkC1,
                       const float* __restrict__ bns, const float* __restrict__ bnt,
                       const float* __restrict__ c2w,
                       float* __restrict__ accum) {
    __shared__ float red[256];
    const int lane   = threadIdx.x & 31;
    const int wave   = threadIdx.x >> 5;
    const int khalf  = lane >> 4;
    const int rowBase = blockIdx.x * 128 + wave * 16;
    const uint4* ap = (const uint4*)(feat + (size_t)(rowBase + (lane & 15)) * FSTRIDE);

    float partial = 0.f;
    for (int nt = 0; nt < 16; ++nt) {
        v8f c = {};
        const unsigned int* bbase = pkC1 + ((size_t)nt * 13) * 256 + lane * 8;
        U8 bu = *(const U8*)bbase;                      // kk = 0
        for (int kk = 0; kk < 13; ++kk) {
            U8 bn;
            if (kk < 12) bn = *(const U8*)(bbase + (size_t)(kk + 1) * 256);
            U8 t;
            *(uint4*)&t.u[0] = ap[kk * 4 + khalf];
            *(uint4*)&t.u[4] = ap[kk * 4 + 2 + khalf];
            v16h a = __builtin_bit_cast(v16h, t);
            v16h b = __builtin_bit_cast(v16h, bu);
            c = __builtin_amdgcn_wmma_f32_16x16x32_f16(
                    false, a, false, b, (short)0, c, false, false);
            if (kk < 12) bu = bn;
        }
        int n = nt * 16 + (lane & 15);
        float sc = bns[n], sh = bnt[n], w = c2w[n];
        #pragma unroll
        for (int i = 0; i < 8; ++i) {
            float h = c[i] * sc + sh;
            h = h > 0.f ? h : 0.f;
            partial += h * w;
        }
    }
    red[threadIdx.x] = partial;
    __syncthreads();
    #pragma unroll
    for (int off = 128; off > 0; off >>= 1) {
        if ((int)threadIdx.x < off) red[threadIdx.x] += red[threadIdx.x + off];
        __syncthreads();
    }
    if (threadIdx.x == 0) atomicAdd(accum + (blockIdx.x >> 5), red[0]);
}

__global__ void gnn_final_kernel(const float* accum, const float* c2b, float* out) {
    int i = threadIdx.x;
    if (i < 64) out[i] = accum[i] * (1.f / 4096.f) + c2b[0];
}

extern "C" void kernel_launch(void* const* d_in, const int* in_sizes, int n_in,
                              void* d_out, int out_size, void* d_ws, size_t ws_size,
                              hipStream_t stream) {
    if (n_in < 22) return;
    const float* x0    = (const float*)d_in[0];   // [64,4096,16]
    const int*   links = (const int*)  d_in[1];   // [64,2,16384]
    const float* W[3]  = {(const float*)d_in[2], (const float*)d_in[6], (const float*)d_in[10]};
    const float* Lb[3] = {(const float*)d_in[3], (const float*)d_in[7], (const float*)d_in[11]};
    const float* Lg[3] = {(const float*)d_in[4], (const float*)d_in[8], (const float*)d_in[12]};
    const float* Lbb[3]= {(const float*)d_in[5], (const float*)d_in[9], (const float*)d_in[13]};
    const float* c1W = (const float*)d_in[14];
    const float* c1b = (const float*)d_in[15];
    const float* bng = (const float*)d_in[16];
    const float* bnb = (const float*)d_in[17];
    const float* bnm = (const float*)d_in[18];
    const float* bnv = (const float*)d_in[19];
    const float* c2W = (const float*)d_in[20];
    const float* c2b = (const float*)d_in[21];

    char* ws = (char*)d_ws;
    float*        accum = (float*)ws;                          //   256 B
    float*        bns   = (float*)(ws + 256);                  //  1 KB
    float*        bnt   = (float*)(ws + 256 + 1024);
    float*        c2w   = (float*)(ws + 256 + 2048);
    unsigned int* pkW1  = (unsigned int*)(ws + 4096);          //   8 KB
    unsigned int* pkW2  = (unsigned int*)(ws + 4096 + 8192);   //  32 KB
    unsigned int* pkW3  = pkW2 + 8192;                         //  32 KB
    unsigned int* pkC1  = pkW3 + 8192;                         // 208 KB
    half_t*       agg   = (half_t*)(ws + 294912);              //  64 MB [rows][<=128] f16
    half_t*       feat  = (half_t*)(ws + 294912 + (size_t)NROWS * 128 * 2); // 208 MB

    gnn_init_kernel<<<NROWS / 256, 256, 0, stream>>>(x0, feat, accum);
    gnn_bnprep_kernel<<<1, 256, 0, stream>>>(c1b, bng, bnb, bnm, bnv, c2W, bns, bnt, c2w);
    gnn_packb_kernel<<< 8,  256, 0, stream>>>(W[0], pkW1,  1,  16, 128);
    gnn_packb_kernel<<<32,  256, 0, stream>>>(W[1], pkW2,  4, 128, 128);
    gnn_packb_kernel<<<32,  256, 0, stream>>>(W[2], pkW3,  4, 128, 128);
    gnn_packb_kernel<<<208, 256, 0, stream>>>(c1W,  pkC1, 13, 400, 256);

    // ---- layer 1: f16 agg stride 32 (K padded 16->32) ----
    gnn_zero_kernel<<<(NROWS * 32 * 2 / 16) / 256, 256, 0, stream>>>((float4*)agg, (long)NROWS * 4);
    gnn_scatter16_kernel<<<(64L * NEDGES * 2) / 256, 256, 0, stream>>>(x0, links, agg);
    gnn_gemm_ln_kernel<1><<<NROWS / 128, 256, 0, stream>>>(agg, (const uint4*)pkW1, Lb[0], Lg[0], Lbb[0], feat, 16);

    // ---- layer 2 ----
    gnn_zero_kernel<<<(NROWS * 128 * 2 / 16) / 256, 256, 0, stream>>>((float4*)agg, (long)NROWS * 16);
    gnn_scatter128_kernel<<<(64L * NEDGES * 16) / 256, 256, 0, stream>>>(feat, 16, links, agg);
    gnn_gemm_ln_kernel<4><<<NROWS / 128, 256, 0, stream>>>(agg, (const uint4*)pkW2, Lb[1], Lg[1], Lbb[1], feat, 144);

    // ---- layer 3 ----
    gnn_zero_kernel<<<(NROWS * 128 * 2 / 16) / 256, 256, 0, stream>>>((float4*)agg, (long)NROWS * 16);
    gnn_scatter128_kernel<<<(64L * NEDGES * 16) / 256, 256, 0, stream>>>(feat, 144, links, agg);
    gnn_gemm_ln_kernel<4><<<NROWS / 128, 256, 0, stream>>>(agg, (const uint4*)pkW3, Lb[2], Lg[2], Lbb[2], feat, 272);

    // ---- critic head + pooling ----
    gnn_critic_kernel<<<NROWS / 128, 256, 0, stream>>>(feat, pkC1, bns, bnt, c2w, accum);
    gnn_final_kernel<<<1, 64, 0, stream>>>(accum, c2b, (float*)d_out);
}
